// SelfAttention_25563645346611
// MI455X (gfx1250) — matrix operand order repaired and checked by reading, not compiled, not run on hardware
//
#include <hip/hip_runtime.h>

#define BATCH 4
#define CH    256
#define NPIX  4096
#define DQK   64
#define WROWS 384   // 64 q-rows + 64 k-rows + 256 v-rows
#define NBLK  (NPIX / 32)

typedef __attribute__((ext_vector_type(16))) _Float16 v16h;
typedef __attribute__((ext_vector_type(8)))  _Float16 v8h;
typedef __attribute__((ext_vector_type(8)))  float    v8f;

// Build a 16-half WMMA fragment register from two 8-half (16B) loads.
__device__ __forceinline__ v16h frag16(const _Float16* lo, const _Float16* hi) {
  v8h a = *(const v8h*)lo;
  v8h b = *(const v8h*)hi;
  v16h r;
#pragma unroll
  for (int i = 0; i < 8; ++i) { r[i] = a[i]; r[i + 8] = b[i]; }
  return r;
}

// CDNA5 async direct-to-LDS copy, 16 bytes. Tracked by ASYNCcnt.
__device__ __forceinline__ void async_b128(const _Float16* lds_dst, const _Float16* g_src) {
  unsigned lds_off = (unsigned)(uintptr_t)lds_dst;           // low 32 bits = LDS byte offset
  unsigned long long ga = (unsigned long long)(uintptr_t)g_src;
  asm volatile("global_load_async_to_lds_b128 %0, %1, off"
               :: "v"(lds_off), "v"(ga) : "memory");
}

// ---------------------------------------------------------------------------
// Kernel 1: pack Wq/Wk/Wv into one f16 [384][256] matrix + packed f32 bias.
// ---------------------------------------------------------------------------
__global__ void pack_weights(const float* __restrict__ Wq, const float* __restrict__ bq,
                             const float* __restrict__ Wk, const float* __restrict__ bk,
                             const float* __restrict__ Wv, const float* __restrict__ bv,
                             _Float16* __restrict__ wall, float* __restrict__ ball) {
  int idx = blockIdx.x * blockDim.x + threadIdx.x;   // 0 .. 384*256-1
  int row = idx >> 8, c = idx & 255;
  float w;
  if (row < 64)       w = Wq[row * CH + c];
  else if (row < 128) w = Wk[(row - 64) * CH + c];
  else                w = Wv[(row - 128) * CH + c];
  wall[idx] = (_Float16)w;
  if (idx < WROWS) {
    float bb;
    if (idx < 64)       bb = bq[idx];
    else if (idx < 128) bb = bk[idx - 64];
    else                bb = bv[idx - 128];
    ball[idx] = bb;
  }
}

// ---------------------------------------------------------------------------
// Kernel 2: projections via WMMA.  Out[384][64px] = Wall[384][256] x xtile.
//   q16: [B][N][64]  k16: [B][N][64]  v16: [B][C][N]   (all f16, +bias)
// ---------------------------------------------------------------------------
#define XT_LD 272   // 256 + 16 halves pad: rows stay 32B aligned, banks spread

__global__ __launch_bounds__(256) void proj_kernel(const float* __restrict__ x,
                                                   const _Float16* __restrict__ wall,
                                                   const float* __restrict__ ball,
                                                   _Float16* __restrict__ q16,
                                                   _Float16* __restrict__ k16,
                                                   _Float16* __restrict__ v16) {
  __shared__ __align__(32) _Float16 xT[64 * XT_LD];   // [pixel][channel] f16
  const int tid = threadIdx.x;
  const int b   = blockIdx.y;
  const int n0  = blockIdx.x * 64;

  // Stage x tile transposed (coalesced global read over n, f32->f16).
  for (int idx = tid; idx < CH * 64; idx += 256) {
    int c = idx >> 6, n = idx & 63;
    xT[n * XT_LD + c] = (_Float16)x[((size_t)b * CH + c) * NPIX + n0 + n];
  }
  __syncthreads();

  const int w = tid >> 5, lane = tid & 31, lh = lane & 15;
  const int hsel = lane >> 4;   // which half-wave

  for (int tt = w; tt < 96; tt += 8) {       // 24 row-tiles x 4 col-tiles
    int rt = tt % 24, ct = tt / 24;
    v8f acc = {};
#pragma unroll
    for (int chnk = 0; chnk < 8; ++chnk) {   // K = 256 in chunks of 32
      // A = Wall rows (M=row=rt*16+lh, K halves: base8 + 0..7, 16+base8+0..7)
      const _Float16* ap = wall + (size_t)(rt * 16 + lh) * CH + chnk * 32 + hsel * 8;
      v16h A = frag16(ap, ap + 16);
      // B = x (K=c halves: hsel*8 + 0..7, 16+hsel*8+0..7; N = pixel = ct*16+lh)
      const _Float16* bp = &xT[(ct * 16 + lh) * XT_LD + chnk * 32 + hsel * 8];
      v16h Bf = frag16(bp, bp + 16);
      acc = __builtin_amdgcn_wmma_f32_16x16x32_f16(false, A, false, Bf,
                                                   (short)0, acc, false, false);
    }
    int n = n0 + ct * 16 + lh;
#pragma unroll
    for (int r = 0; r < 8; ++r) {            // C layout: row = r + 8*hsel
      int row = rt * 16 + r + hsel * 8;
      _Float16 hv = (_Float16)(acc[r] + ball[row]);
      if (row < 64)       q16[((size_t)b * NPIX + n) * DQK + row] = hv;
      else if (row < 128) k16[((size_t)b * NPIX + n) * DQK + (row - 64)] = hv;
      else                v16[((size_t)b * CH + (row - 128)) * NPIX + n] = hv;
    }
  }
}

// ---------------------------------------------------------------------------
// Kernel 3: flash attention.  8 waves x 16 queries per WG.
//   S^T tile trick: C-layout of S^T == A-layout of P for the O WMMA.
//   Double-buffered CDNA5 async global->LDS staging (ASYNCcnt).
// ---------------------------------------------------------------------------
#define KL 72    // 64 + 8 halves pad (144B rows, 16B aligned)
#define VL 40    // 32 + 8 halves pad (80B rows, 16B aligned)

__global__ __launch_bounds__(256) void attn_kernel(const float* __restrict__ x,
                                                   const float* __restrict__ gamma,
                                                   const _Float16* __restrict__ q16,
                                                   const _Float16* __restrict__ k16,
                                                   const _Float16* __restrict__ v16,
                                                   float* __restrict__ out) {
  __shared__ __align__(32) _Float16 k_lds[2][32 * KL];   // [buf][key][d]
  __shared__ __align__(32) _Float16 v_lds[2][CH * VL];   // [buf][channel][key]
  const int tid = threadIdx.x;
  const int b   = blockIdx.y;
  const int w = tid >> 5, lane = tid & 31, lh = lane & 15, hsel = lane >> 4;
  const int qbase = blockIdx.x * 128 + w * 16;

  // Per-thread staging sources (5 async b128 per block: 1 K + 4 V).
  const int skey = tid >> 3, sdg = tid & 7;
  const _Float16* kin = k16 + ((size_t)b * NPIX + skey) * DQK + sdg * 8;
  const _Float16* vin = v16 + ((size_t)b * CH + tid) * NPIX;

  // Q^T as B-fragments: N=query=lh, K(d) = chunk*32 + hsel*16 + i (contiguous).
  v16h Qf[2];
#pragma unroll
  for (int chunk = 0; chunk < 2; ++chunk) {
    const _Float16* qp = q16 + ((size_t)b * NPIX + qbase + lh) * DQK + chunk * 32 + hsel * 16;
    Qf[chunk] = frag16(qp, qp + 8);
  }

  v8f acc[16];                       // O: 16 queries x 256 channels, f32
#pragma unroll
  for (int i = 0; i < 16; ++i) acc[i] = (v8f){};
  float m_run = -3.0e38f, l_run = 0.f;

  // Prologue: stage block 0 into buffer 0.
  async_b128(&k_lds[0][skey * KL + sdg * 8], kin);
#pragma unroll
  for (int g = 0; g < 4; ++g)
    async_b128(&v_lds[0][tid * VL + g * 8], vin + g * 8);

  for (int kb = 0; kb < NBLK; ++kb) {
    const int cur = kb & 1;
    __syncthreads();   // everyone done computing on buf[!cur] before overwrite
    if (kb + 1 < NBLK) {
      const int j1 = (kb + 1) * 32;
      async_b128(&k_lds[cur ^ 1][skey * KL + sdg * 8], kin + (size_t)j1 * DQK);
#pragma unroll
      for (int g = 0; g < 4; ++g)
        async_b128(&v_lds[cur ^ 1][tid * VL + g * 8], vin + j1 + g * 8);
      // block kb's 5 loads retire in issue order: <=5 outstanding => they're done
      asm volatile("s_wait_asynccnt 0x5" ::: "memory");
    } else {
      asm volatile("s_wait_asynccnt 0x0" ::: "memory");
    }
    __syncthreads();   // all waves' block-kb data visible in LDS

    // S^T tiles (M=key, N=query): A = K rows from LDS, B = Q^T regs.
    v8f s[2];
#pragma unroll
    for (int kt = 0; kt < 2; ++kt) {
      v8f cacc = (v8f){};
#pragma unroll
      for (int chunk = 0; chunk < 2; ++chunk) {
        const _Float16* kp = &k_lds[cur][(kt * 16 + lh) * KL + chunk * 32 + hsel * 16];
        v16h A = frag16(kp, kp + 8);
        cacc = __builtin_amdgcn_wmma_f32_16x16x32_f16(false, A, false, Qf[chunk],
                                                      (short)0, cacc, false, false);
      }
      s[kt] = cacc;
    }

    // Online softmax: per lane, fixed query lh, keys r + 8*hsel (+16 for s[1]).
    float mx = -3.0e38f;
#pragma unroll
    for (int r = 0; r < 8; ++r) { mx = fmaxf(mx, s[0][r]); mx = fmaxf(mx, s[1][r]); }
    mx = fmaxf(mx, __shfl_xor(mx, 16));
    float m_new   = fmaxf(m_run, mx);
    float rescale = __expf(m_run - m_new);
    float psum = 0.f;
    v16h Pf;   // S^T C-layout == P A-layout: pack exp() directly, no permutes
#pragma unroll
    for (int r = 0; r < 8; ++r) {
      float p0 = __expf(s[0][r] - m_new);
      float p1 = __expf(s[1][r] - m_new);
      psum += p0 + p1;
      Pf[r]     = (_Float16)p0;
      Pf[r + 8] = (_Float16)p1;
    }
    psum += __shfl_xor(psum, 16);
    l_run = l_run * rescale + psum;
    m_run = m_new;

    // Rescale O accumulator: per-vgpr-row query = r + 8*hsel, stats live in lane q.
    float sc[8];
#pragma unroll
    for (int r = 0; r < 8; ++r) sc[r] = __shfl(rescale, r + hsel * 8);
#pragma unroll
    for (int ct = 0; ct < 16; ++ct)
#pragma unroll
      for (int r = 0; r < 8; ++r) acc[ct][r] *= sc[r];

    // O += P[16x32] x V^T[32x16] per channel tile.
#pragma unroll
    for (int ct = 0; ct < 16; ++ct) {
      const _Float16* vp = &v_lds[cur][(ct * 16 + lh) * VL + hsel * 8];
      v16h Bf = frag16(vp, vp + 16);
      acc[ct] = __builtin_amdgcn_wmma_f32_16x16x32_f16(false, Pf, false, Bf,
                                                       (short)0, acc[ct], false, false);
    }
  }

  // Epilogue: out[b][c][n] = gamma * O/l + x
  float li[8];
#pragma unroll
  for (int r = 0; r < 8; ++r) li[r] = __shfl(l_run, r + hsel * 8);
  const float g = gamma[0];
#pragma unroll
  for (int ct = 0; ct < 16; ++ct) {
#pragma unroll
    for (int r = 0; r < 8; ++r) {
      int c = ct * 16 + lh;
      int n = qbase + r + hsel * 8;
      size_t idx = ((size_t)b * CH + c) * NPIX + n;
      out[idx] = g * (acc[ct][r] / li[r]) + x[idx];
    }
  }
}

// ---------------------------------------------------------------------------
extern "C" void kernel_launch(void* const* d_in, const int* in_sizes, int n_in,
                              void* d_out, int out_size, void* d_ws, size_t ws_size,
                              hipStream_t stream) {
  (void)in_sizes; (void)n_in; (void)out_size; (void)ws_size;
  const float* x     = (const float*)d_in[0];
  const float* Wq    = (const float*)d_in[1];
  const float* bq    = (const float*)d_in[2];
  const float* Wk    = (const float*)d_in[3];
  const float* bk    = (const float*)d_in[4];
  const float* Wv    = (const float*)d_in[5];
  const float* bv    = (const float*)d_in[6];
  const float* gamma = (const float*)d_in[7];
  float* out = (float*)d_out;

  char* ws = (char*)d_ws;
  size_t off = 0;
  auto alloc = [&](size_t bytes) -> void* {
    void* p = ws + off;
    off += (bytes + 255) & ~(size_t)255;
    return p;
  };
  _Float16* wall = (_Float16*)alloc((size_t)WROWS * CH * sizeof(_Float16));
  float*    ball = (float*)alloc((size_t)WROWS * sizeof(float));
  _Float16* q16  = (_Float16*)alloc((size_t)BATCH * NPIX * DQK * sizeof(_Float16));
  _Float16* k16  = (_Float16*)alloc((size_t)BATCH * NPIX * DQK * sizeof(_Float16));
  _Float16* v16  = (_Float16*)alloc((size_t)BATCH * CH * NPIX * sizeof(_Float16));

  pack_weights<<<dim3((WROWS * CH) / 256), dim3(256), 0, stream>>>(
      Wq, bq, Wk, bk, Wv, bv, wall, ball);
  proj_kernel<<<dim3(NPIX / 64, BATCH), dim3(256), 0, stream>>>(
      x, wall, ball, q16, k16, v16);
  attn_kernel<<<dim3(NPIX / 128, BATCH), dim3(256), 0, stream>>>(
      x, gamma, q16, k16, v16, out);
}
